// AttnRNNLM_27547920237229
// MI455X (gfx1250) — matrix operand order repaired and checked by reading, not compiled, not run on hardware
//
#include <hip/hip_runtime.h>

// ---------------------------------------------------------------------------
// Problem constants (match reference)
// ---------------------------------------------------------------------------
#define VOCAB 32000
#define EDIM  512
#define HDIM  1024
#define BATCH 4
#define SEQ   1024
#define MROWS (BATCH * SEQ)   // 4096

// ---------------------------------------------------------------------------
// WMMA fragment helpers (gfx1250 wave32, v_wmma_f32_16x16x32_bf16)
// ---------------------------------------------------------------------------
typedef __attribute__((ext_vector_type(16))) __bf16 bf16x16;
typedef __attribute__((ext_vector_type(8)))  float  floatx8;
typedef __attribute__((ext_vector_type(4)))  unsigned uint32x4;
typedef __attribute__((ext_vector_type(8)))  unsigned uint32x8;

union FragBF { bf16x16 v; unsigned u[8]; };

__device__ __forceinline__ floatx8 wmma_bf16(bf16x16 a, bf16x16 b, floatx8 c) {
  return __builtin_amdgcn_wmma_f32_16x16x32_bf16(
      /*neg_a=*/false, a, /*neg_b=*/false, b,
      /*c_mod=*/(short)0, c, /*reuse_a=*/false, /*reuse_b=*/false);
}

// A-matrix 16x32 bf16 fragment from row-major [16][ld] memory (LDS or global).
// ISA 7.12.2: lane = half*16 + m ; VGPR j<4 -> K = 2j + 8*half ; j>=4 -> K = 16+2(j-4)+8*half
__device__ __forceinline__ bf16x16 load_a_frag(const __bf16* base, int ld) {
  const int lane = threadIdx.x & 31;
  const int half = lane >> 4;
  const int m    = lane & 15;
  const unsigned* rowp = reinterpret_cast<const unsigned*>(base + (long long)m * ld);
  FragBF f;
#pragma unroll
  for (int j = 0; j < 8; ++j) {
    const int kbase = (j < 4 ? 2 * j : 16 + 2 * (j - 4)) + 8 * half;
    f.u[j] = rowp[kbase >> 1];   // two consecutive bf16 (K, K+1)
  }
  return f.v;
}

// B-matrix 32x16 bf16 fragment from row-major [32][ld] memory: lane l holds column
// N=l%16; element s -> K = 8*half + s (s<8), 16 + 8*half + (s-8) (s>=8).
__device__ __forceinline__ bf16x16 load_b_frag(const __bf16* base, int ld) {
  const int lane = threadIdx.x & 31;
  const int half = lane >> 4;
  const int n    = lane & 15;
  bf16x16 v;
#pragma unroll
  for (int s = 0; s < 16; ++s) {
    const int k = (s < 8 ? s : 8 + s) + 8 * half;
    v[s] = base[(long long)k * ld + n];
  }
  return v;
}

// ---------------------------------------------------------------------------
// Tensor Data Mover: asynchronous 2-D tile DMA  global -> LDS.
// Builds a D# (groups 0+1, 2-D tensor form, data_size = 2 bytes) in SGPRs and
// issues TENSOR_LOAD_TO_LDS via inline asm (ISA ch.7/8 of async_tensor doc).
// Tracked by TENSORcnt; completion enforced with s_wait_tensorcnt.
//   tile_d0 = contiguous elements per row, tile_d1 = rows
//   stride0 = tensor_dim0_stride (elements between consecutive rows)
// LDS result is compact row-major [tile_d1][tile_d0].
// ---------------------------------------------------------------------------
__device__ __forceinline__ void tdm_load_2d(unsigned lds_addr, const void* gaddr,
                                            unsigned tensor_d0, unsigned tensor_d1,
                                            unsigned tile_d0, unsigned tile_d1,
                                            unsigned long long stride0) {
  const unsigned long long ga = (unsigned long long)gaddr;
  uint32x4 g0;
  g0[0] = 1u;                                   // count=1, user descriptor, no gather
  g0[1] = lds_addr;                             // LDS byte address
  g0[2] = (unsigned)ga;                         // global_addr[31:0]
  g0[3] = (unsigned)((ga >> 32) & 0x01FFFFFFu)  // global_addr[56:32]
          | (2u << 30);                         // type = 2 ("image")
  uint32x8 g1;
  g1[0] = 1u << 16;                             // workgroup_mask=0, data_size=1 (2B)
  g1[1] = (tensor_d0 & 0xFFFFu) << 16;          // tensor_dim0[15:0] at bits 63:48
  g1[2] = (tensor_d0 >> 16) | ((tensor_d1 & 0xFFFFu) << 16);
  g1[3] = (tensor_d1 >> 16) | (tile_d0 << 16);  // tile_dim0 at bits 127:112
  g1[4] = tile_d1;                              // tile_dim1; tile_dim2 = 0 (2-D)
  g1[5] = (unsigned)stride0;                    // tensor_dim0_stride[31:0]
  g1[6] = (unsigned)(stride0 >> 32) & 0xFFFFu;  // stride[47:32]; dim1_stride=0
  g1[7] = 0u;
  asm volatile("tensor_load_to_lds %0, %1" :: "s"(g0), "s"(g1) : "memory");
}

__device__ __forceinline__ unsigned lds_byte_offset(const void* p) {
  // Generic LDS address: low 32 bits are the LDS byte offset (ISA 10.2).
  return (unsigned)(unsigned long long)p;
}

// ---------------------------------------------------------------------------
// Elementwise conversion / gather / concat / transpose kernels
// ---------------------------------------------------------------------------
__global__ void cvt_f32_bf16_kernel(const float* __restrict__ src,
                                    __bf16* __restrict__ dst, long long n) {
  long long i = (long long)blockIdx.x * 256 + threadIdx.x;
  const long long stride = (long long)gridDim.x * 256;
  for (; i < n; i += stride) dst[i] = (__bf16)src[i];
}

__global__ void gather_embed_kernel(const int* __restrict__ x,
                                    const float* __restrict__ emb,
                                    __bf16* __restrict__ e) {
  const long long r = blockIdx.x;            // 0 .. MROWS-1 (b*T + t)
  const long long row = x[r];
  for (int i = threadIdx.x; i < EDIM; i += blockDim.x)
    e[r * EDIM + i] = (__bf16)emb[row * EDIM + i];
}

__global__ void concat_kernel(const __bf16* __restrict__ hs,
                              const __bf16* __restrict__ ctx,
                              __bf16* __restrict__ cat) {
  const long long r = blockIdx.x;
  for (int i = threadIdx.x; i < HDIM; i += blockDim.x) {
    cat[r * (2 * HDIM) + i]        = hs[r * HDIM + i];
    cat[r * (2 * HDIM) + HDIM + i] = ctx[r * HDIM + i];
  }
}

// 32x32 LDS tile transpose: dst[h][s] = src[s][h]  (per batch via blockIdx.z)
__global__ __launch_bounds__(256)
void transpose_bf16_kernel(const __bf16* __restrict__ src,   // [SEQ, HDIM]
                           __bf16* __restrict__ dst) {       // [HDIM, SEQ]
  __shared__ __bf16 tile[32][33];   // +1 pad: avoid bank conflicts
  const long long bz = blockIdx.z;
  const __bf16* s = src + bz * (long long)SEQ * HDIM;
  __bf16* d = dst + bz * (long long)SEQ * HDIM;
  const int s0 = blockIdx.y * 32;           // seq base
  const int h0 = blockIdx.x * 32;           // hidden base
  const int lx = threadIdx.x & 31, ly = threadIdx.x >> 5;   // 32 x 8
#pragma unroll
  for (int r = ly; r < 32; r += 8)
    tile[r][lx] = s[(long long)(s0 + r) * HDIM + h0 + lx];
  __syncthreads();
#pragma unroll
  for (int r = ly; r < 32; r += 8)
    d[(long long)(h0 + r) * SEQ + s0 + lx] = tile[lx][r];
}

// ---------------------------------------------------------------------------
// Tiled bf16 WMMA GEMM with TDM-staged, double-buffered LDS tiles:
//   C[M,N] = act(scale * A[M,K] @ B[K,N] + bias)
// Batched over blockIdx.z with element strides sA/sB/sC.
// Block: 256 threads (8 waves). Tile: 128x128, k-step 32.
// Wave 0 drives the Tensor Data Mover: while waves compute tile i, the TDM
// streams tile i+1 into the other LDS buffer (TENSORcnt sync, in-order).
// All M,N,K here are multiples of 128/128/32 -> no edge guards.
// ---------------------------------------------------------------------------
#define BM 128
#define BN 128
#define BK 32

template <bool HAS_BIAS, bool DO_TANH, bool OUT_BF16>
__global__ __launch_bounds__(256)
void gemm_bf16_kernel(const __bf16* __restrict__ Aall,
                      const __bf16* __restrict__ Ball,
                      const float* __restrict__ bias,
                      void* __restrict__ Call,
                      int M, int N, int K, float scale,
                      long long sA, long long sB, long long sC) {
  __shared__ __align__(16) __bf16 Asm[2][BM][BK];
  __shared__ __align__(16) __bf16 Bsm[2][BK][BN];

  const __bf16* A = Aall + sA * blockIdx.z;
  const __bf16* B = Ball + sB * blockIdx.z;

  const int m0 = blockIdx.y * BM;
  const int n0 = blockIdx.x * BN;

  const int tid  = threadIdx.x;
  const int wave = tid >> 5;
  const int lane = tid & 31;
  const int wm   = wave >> 2;   // 0..1
  const int wn   = wave & 3;    // 0..3

  const unsigned ldsA[2] = {lds_byte_offset(&Asm[0][0][0]), lds_byte_offset(&Asm[1][0][0])};
  const unsigned ldsB[2] = {lds_byte_offset(&Bsm[0][0][0]), lds_byte_offset(&Bsm[1][0][0])};

  const floatx8 zero = {0.f, 0.f, 0.f, 0.f, 0.f, 0.f, 0.f, 0.f};
  floatx8 acc[4][2];
#pragma unroll
  for (int i = 0; i < 4; ++i)
#pragma unroll
    for (int j = 0; j < 2; ++j) acc[i][j] = zero;

  const int nk = K / BK;
  // Prologue: TDM pair for tile 0 into buffer 0 (wave 0 only; EXEC-independent DMA)
  if (tid < 32) {
    tdm_load_2d(ldsA[0], A + (long long)m0 * K, (unsigned)K, (unsigned)M,
                BK, BM, (unsigned long long)K);
    tdm_load_2d(ldsB[0], B + (long long)0 * N + n0, (unsigned)N, (unsigned)K,
                BN, BK, (unsigned long long)N);
  }

  for (int i = 0; i < nk; ++i) {
    const int buf = i & 1;
    if (tid < 32) {
      if (i + 1 < nk) {
        // Prefetch tile i+1 via TDM into the other buffer, then wait for tile i
        // (TDM ops of one wave complete in order; <=2 outstanding == pair i done).
        tdm_load_2d(ldsA[buf ^ 1], A + (long long)m0 * K + (long long)(i + 1) * BK,
                    (unsigned)K, (unsigned)M, BK, BM, (unsigned long long)K);
        tdm_load_2d(ldsB[buf ^ 1], B + (long long)(i + 1) * BK * N + n0,
                    (unsigned)N, (unsigned)K, BN, BK, (unsigned long long)N);
        __builtin_amdgcn_s_wait_tensorcnt(2);
      } else {
        __builtin_amdgcn_s_wait_tensorcnt(0);
      }
    }
    __syncthreads();   // tile i visible to all waves

    // ---- compute: 8 WMMAs per wave per k-step ----
    bf16x16 bfr0 = load_b_frag(&Bsm[buf][0][wn * 32], BN);
    bf16x16 bfr1 = load_b_frag(&Bsm[buf][0][wn * 32 + 16], BN);
#pragma unroll
    for (int mf = 0; mf < 4; ++mf) {
      bf16x16 afr = load_a_frag(&Asm[buf][wm * 64 + mf * 16][0], BK);
      acc[mf][0] = wmma_bf16(afr, bfr0, acc[mf][0]);
      acc[mf][1] = wmma_bf16(afr, bfr1, acc[mf][1]);
    }
    __syncthreads();   // done reading buf before TDM refills it next iteration
  }

  // ---- epilogue: C/D layout row = reg + 8*half, col = lane&15 ----
  const int half = lane >> 4;
  const int ncol = lane & 15;
#pragma unroll
  for (int mf = 0; mf < 4; ++mf) {
#pragma unroll
    for (int nf = 0; nf < 2; ++nf) {
      const int col = n0 + wn * 32 + nf * 16 + ncol;
      const float bv = HAS_BIAS ? bias[col] : 0.f;
#pragma unroll
      for (int r = 0; r < 8; ++r) {
        const int row = m0 + wm * 64 + mf * 16 + r + 8 * half;
        float v = acc[mf][nf][r] * scale + bv;
        if (DO_TANH) v = tanhf(v);
        const long long idx = sC * blockIdx.z + (long long)row * N + col;
        if (OUT_BF16) ((__bf16*)Call)[idx] = (__bf16)v;
        else          ((float*)Call)[idx]  = v;
      }
    }
  }
}

// ---------------------------------------------------------------------------
// Sequential RNN recurrence: h_t = tanh(pre_t + h_{t-1} @ w_hh)
// One workgroup, 1024 threads = 32 waves. h kept in LDS as bf16, padded to
// M=16 rows (batches 0..3 live, 4..15 stay zero) -> whole state is a single
// A-fragment row. Wave w owns 32 output columns (2 fragments); w_hh (2 MB)
// streams from L2 every step.
// ---------------------------------------------------------------------------
__global__ __launch_bounds__(1024)
void rnn_kernel(const float* __restrict__ pre, const __bf16* __restrict__ whh,
                __bf16* __restrict__ hs) {
  __shared__ __align__(16) __bf16 hstate[16][HDIM];

  const int tid = threadIdx.x;
  for (int i = tid; i < 16 * HDIM; i += 1024)
    (&hstate[0][0])[i] = (__bf16)0.f;
  __syncthreads();

  const int wave = tid >> 5;
  const int lane = tid & 31;
  const int nb   = wave * 32;
  const int half = lane >> 4;
  const int ncol = lane & 15;
  const floatx8 zero = {0.f, 0.f, 0.f, 0.f, 0.f, 0.f, 0.f, 0.f};

  for (int t = 0; t < SEQ; ++t) {
    floatx8 acc0 = zero, acc1 = zero;
    for (int k0 = 0; k0 < HDIM; k0 += 32) {
      bf16x16 a  = load_a_frag(&hstate[0][k0], HDIM);                      // LDS
      bf16x16 b0 = load_b_frag(whh + (long long)k0 * HDIM + nb, HDIM);     // L2-resident
      bf16x16 b1 = load_b_frag(whh + (long long)k0 * HDIM + nb + 16, HDIM);
      acc0 = wmma_bf16(a, b0, acc0);
      acc1 = wmma_bf16(a, b1, acc1);
    }
    __syncthreads();   // all waves done reading hstate for step t
    if (half == 0) {
#pragma unroll
      for (int r = 0; r < 4; ++r) {      // rows r = batch index
        const long long row = (long long)r * SEQ + t;
        const float v0 = tanhf(acc0[r] + pre[row * HDIM + nb + ncol]);
        const float v1 = tanhf(acc1[r] + pre[row * HDIM + nb + 16 + ncol]);
        const __bf16 h0 = (__bf16)v0, h1 = (__bf16)v1;
        hstate[r][nb + ncol]      = h0;
        hstate[r][nb + 16 + ncol] = h1;
        hs[row * HDIM + nb + ncol]      = h0;
        hs[row * HDIM + nb + 16 + ncol] = h1;
      }
    }
    __syncthreads();   // hstate updated before step t+1
  }
}

// ---------------------------------------------------------------------------
// Causal softmax over pre-scaled scores [B*T, T] (f32 in, bf16 probs out)
// ---------------------------------------------------------------------------
__device__ __forceinline__ float block_reduce(float v, float* tmp, bool do_max) {
#pragma unroll
  for (int off = 16; off > 0; off >>= 1) {
    const float o = __shfl_xor(v, off, 32);
    v = do_max ? fmaxf(v, o) : (v + o);
  }
  const int wid = threadIdx.x >> 5, lane = threadIdx.x & 31;
  if (lane == 0) tmp[wid] = v;
  __syncthreads();
  v = (lane < 8) ? tmp[lane] : (do_max ? -3.0e38f : 0.f);
  if (wid == 0) {
#pragma unroll
    for (int off = 4; off > 0; off >>= 1) {
      const float o = __shfl_xor(v, off, 32);
      v = do_max ? fmaxf(v, o) : (v + o);
    }
    if (lane == 0) tmp[0] = v;
  }
  __syncthreads();
  v = tmp[0];
  __syncthreads();
  return v;
}

__global__ __launch_bounds__(256)
void softmax_causal_kernel(const float* __restrict__ scores,
                           __bf16* __restrict__ probs) {
  __shared__ float tmp[8];
  const long long row = blockIdx.x;              // b*T + t
  const int t   = (int)(row % SEQ);
  const int len = t + 1;
  const float* srow = scores + row * SEQ;
  __bf16* prow = probs + row * SEQ;

  float mx = -3.0e38f;
  for (int s = threadIdx.x; s < len; s += blockDim.x) mx = fmaxf(mx, srow[s]);
  mx = block_reduce(mx, tmp, true);

  float sum = 0.f;
  for (int s = threadIdx.x; s < len; s += blockDim.x) sum += expf(srow[s] - mx);
  sum = block_reduce(sum, tmp, false);
  const float inv = 1.f / sum;

  for (int s = threadIdx.x; s < SEQ; s += blockDim.x)
    prow[s] = (__bf16)(s < len ? expf(srow[s] - mx) * inv : 0.f);
}

// ---------------------------------------------------------------------------
// Host-side launcher
// ---------------------------------------------------------------------------
static inline unsigned cvt_grid(long long n) {
  long long g = (n + 255) / 256;
  return (unsigned)(g > 4096 ? 4096 : g);
}

extern "C" void kernel_launch(void* const* d_in, const int* in_sizes, int n_in,
                              void* d_out, int out_size, void* d_ws, size_t ws_size,
                              hipStream_t stream) {
  // inputs (setup_inputs order)
  const int*   x      = (const int*)  d_in[0];
  const float* emb    = (const float*)d_in[1];
  const float* w_ih   = (const float*)d_in[2];
  const float* b_ih   = (const float*)d_in[3];
  const float* w_hh   = (const float*)d_in[4];
  const float* wq     = (const float*)d_in[5];
  const float* wk     = (const float*)d_in[6];
  const float* wv     = (const float*)d_in[7];
  const float* w_mix  = (const float*)d_in[8];
  const float* b_mix  = (const float*)d_in[9];
  const float* w_proj = (const float*)d_in[10];
  const float* b_proj = (const float*)d_in[11];
  float* out = (float*)d_out;

  // workspace carve-up (256B aligned)
  char* wp = (char*)d_ws;
  auto alloc = [&](size_t bytes) {
    char* p = wp;
    wp += (bytes + 255) & ~(size_t)255;
    return p;
  };
  __bf16* e_b     = (__bf16*)alloc((size_t)MROWS * EDIM * 2);
  __bf16* w_ih_b  = (__bf16*)alloc((size_t)EDIM * HDIM * 2);
  float*  pre     = (float*) alloc((size_t)MROWS * HDIM * 4);
  __bf16* w_hh_b  = (__bf16*)alloc((size_t)HDIM * HDIM * 2);
  __bf16* hs_b    = (__bf16*)alloc((size_t)MROWS * HDIM * 2);
  __bf16* wq_b    = (__bf16*)alloc((size_t)HDIM * HDIM * 2);
  __bf16* wk_b    = (__bf16*)alloc((size_t)HDIM * HDIM * 2);
  __bf16* wv_b    = (__bf16*)alloc((size_t)HDIM * HDIM * 2);
  __bf16* q_b     = (__bf16*)alloc((size_t)MROWS * HDIM * 2);
  __bf16* k_b     = (__bf16*)alloc((size_t)MROWS * HDIM * 2);
  __bf16* v_b     = (__bf16*)alloc((size_t)MROWS * HDIM * 2);
  __bf16* kT_b    = (__bf16*)alloc((size_t)MROWS * HDIM * 2);   // per-batch [H, T]
  float*  scores  = (float*) alloc((size_t)BATCH * SEQ * SEQ * 4);
  __bf16* probs   = (__bf16*)alloc((size_t)BATCH * SEQ * SEQ * 2);
  __bf16* ctx_b   = (__bf16*)alloc((size_t)MROWS * HDIM * 2);
  __bf16* cat_b   = (__bf16*)alloc((size_t)MROWS * 2 * HDIM * 2);
  __bf16* w_mix_b = (__bf16*)alloc((size_t)2 * HDIM * HDIM * 2);
  __bf16* h2_b    = (__bf16*)alloc((size_t)MROWS * HDIM * 2);
  __bf16* w_pj_b  = (__bf16*)alloc((size_t)HDIM * VOCAB * 2);

  // 1) weight conversions f32 -> bf16
  cvt_f32_bf16_kernel<<<cvt_grid((long long)EDIM * HDIM), 256, 0, stream>>>(w_ih, w_ih_b, (long long)EDIM * HDIM);
  cvt_f32_bf16_kernel<<<cvt_grid((long long)HDIM * HDIM), 256, 0, stream>>>(w_hh, w_hh_b, (long long)HDIM * HDIM);
  cvt_f32_bf16_kernel<<<cvt_grid((long long)HDIM * HDIM), 256, 0, stream>>>(wq, wq_b, (long long)HDIM * HDIM);
  cvt_f32_bf16_kernel<<<cvt_grid((long long)HDIM * HDIM), 256, 0, stream>>>(wk, wk_b, (long long)HDIM * HDIM);
  cvt_f32_bf16_kernel<<<cvt_grid((long long)HDIM * HDIM), 256, 0, stream>>>(wv, wv_b, (long long)HDIM * HDIM);
  cvt_f32_bf16_kernel<<<cvt_grid((long long)2 * HDIM * HDIM), 256, 0, stream>>>(w_mix, w_mix_b, (long long)2 * HDIM * HDIM);
  cvt_f32_bf16_kernel<<<cvt_grid((long long)HDIM * VOCAB), 256, 0, stream>>>(w_proj, w_pj_b, (long long)HDIM * VOCAB);

  // 2) embedding gather (+ bf16 convert)
  gather_embed_kernel<<<MROWS, 256, 0, stream>>>(x, emb, e_b);

  // 3) pre = e @ w_ih + b_ih   [4096,512]@[512,1024] -> f32
  gemm_bf16_kernel<true, false, false>
      <<<dim3(HDIM / BN, MROWS / BM, 1), 256, 0, stream>>>(
          e_b, w_ih_b, b_ih, pre, MROWS, HDIM, EDIM, 1.f, 0, 0, 0);

  // 4) sequential RNN -> hs (bf16)
  rnn_kernel<<<1, 1024, 0, stream>>>(pre, w_hh_b, hs_b);

  // 5) q,k,v = hs @ wq/wk/wv
  gemm_bf16_kernel<false, false, true>
      <<<dim3(HDIM / BN, MROWS / BM, 1), 256, 0, stream>>>(
          hs_b, wq_b, nullptr, q_b, MROWS, HDIM, HDIM, 1.f, 0, 0, 0);
  gemm_bf16_kernel<false, false, true>
      <<<dim3(HDIM / BN, MROWS / BM, 1), 256, 0, stream>>>(
          hs_b, wk_b, nullptr, k_b, MROWS, HDIM, HDIM, 1.f, 0, 0, 0);
  gemm_bf16_kernel<false, false, true>
      <<<dim3(HDIM / BN, MROWS / BM, 1), 256, 0, stream>>>(
          hs_b, wv_b, nullptr, v_b, MROWS, HDIM, HDIM, 1.f, 0, 0, 0);

  // 6a) kT_b = k_b^T per batch (so the scores GEMM is NN and TDM-staged)
  transpose_bf16_kernel<<<dim3(HDIM / 32, SEQ / 32, BATCH), 256, 0, stream>>>(k_b, kT_b);

  // 6b) scores_b = (q_b @ kT_b) / sqrt(H)   batched over B via grid.z
  gemm_bf16_kernel<false, false, false>
      <<<dim3(SEQ / BN, SEQ / BM, BATCH), 256, 0, stream>>>(
          q_b, kT_b, nullptr, scores, SEQ, SEQ, HDIM, 0.03125f,
          (long long)SEQ * HDIM, (long long)SEQ * HDIM, (long long)SEQ * SEQ);

  // 7) causal softmax -> probs (bf16)
  softmax_causal_kernel<<<MROWS, 256, 0, stream>>>(scores, probs);

  // 8) ctx_b = probs_b @ v_b   batched
  gemm_bf16_kernel<false, false, true>
      <<<dim3(HDIM / BN, SEQ / BM, BATCH), 256, 0, stream>>>(
          probs, v_b, nullptr, ctx_b, SEQ, HDIM, SEQ, 1.f,
          (long long)SEQ * SEQ, (long long)SEQ * HDIM, (long long)SEQ * HDIM);

  // 9) cat = [hs, ctx]
  concat_kernel<<<MROWS, 256, 0, stream>>>(hs_b, ctx_b, cat_b);

  // 10) h2 = tanh(cat @ w_mix + b_mix)
  gemm_bf16_kernel<true, true, true>
      <<<dim3(HDIM / BN, MROWS / BM, 1), 256, 0, stream>>>(
          cat_b, w_mix_b, b_mix, h2_b, MROWS, HDIM, 2 * HDIM, 1.f, 0, 0, 0);

  // 11) out = h2 @ w_proj + b_proj  (dominant GEMM: 4096x32000x1024)
  gemm_bf16_kernel<true, false, false>
      <<<dim3(VOCAB / BN, MROWS / BM, 1), 256, 0, stream>>>(
          h2_b, w_pj_b, b_proj, out, MROWS, VOCAB, HDIM, 1.f, 0, 0, 0);
}
